// GCN_68796786147746
// MI455X (gfx1250) — matrix operand order repaired and checked by reading, not compiled, not run on hardware
//
#include <hip/hip_runtime.h>
#include <hip/hip_bf16.h>

#define F 128
#define NEG_SLOPE 0.2f

typedef __attribute__((ext_vector_type(16))) _Float16 v16h;
typedef __attribute__((ext_vector_type(8)))  _Float16 v8h;
typedef __attribute__((ext_vector_type(8)))  float    v8f;

// ---------------- degree kernels ----------------
__global__ void init_deg_kernel(float* __restrict__ deg, int n) {
  int i = blockIdx.x * blockDim.x + threadIdx.x;
  if (i < n) deg[i] = 1.0f;  // self-loop
}

__global__ void count_deg_kernel(const int* __restrict__ dst, float* __restrict__ deg, int e) {
  int i = blockIdx.x * blockDim.x + threadIdx.x;
  if (i < e) atomicAdd(&deg[dst[i]], 1.0f);
}

__global__ void finalize_deg_kernel(const float* __restrict__ deg,
                                    float* __restrict__ isd, float* __restrict__ invd, int n) {
  int i = blockIdx.x * blockDim.x + threadIdx.x;
  if (i < n) {
    float d = deg[i];
    isd[i]  = rsqrtf(d);
    invd[i] = 1.0f / d;
  }
}

// ---------------- precision / layout conversion ----------------
__global__ void f32_to_f16_kernel(const float* __restrict__ in, _Float16* __restrict__ out, int total) {
  int i = blockIdx.x * blockDim.x + threadIdx.x;
  if (i < total) out[i] = (_Float16)in[i];
}

// Wt[c][k] = (f16) W[k][c]  (so a B fragment is a contiguous K-run)
__global__ void convert_transpose_w_kernel(const float* __restrict__ Wsrc, _Float16* __restrict__ Wt) {
  int i = blockIdx.x * blockDim.x + threadIdx.x;
  if (i < F * F) {
    int k = i / F, c = i % F;
    Wt[(size_t)c * F + k] = (_Float16)Wsrc[i];
  }
}

__global__ void leaky_to_half_kernel(const float* __restrict__ in, _Float16* __restrict__ out, int total) {
  int i = blockIdx.x * blockDim.x + threadIdx.x;
  if (i < total) {
    float v = in[i];
    v = v > 0.0f ? v : NEG_SLOPE * v;
    out[i] = (_Float16)v;
  }
}

// ---------------- WMMA GEMM: Hout = Ah @ Bt^T ; Agg = Hout*invd + bias ----------------
// Ah: [n x 128] f16 row-major.  Bt: [128 x 128] f16, Bt[c][k] = W[k][c].
// block = 256 thr = 8 waves; block b computes rows [16b,16b+16) x all 128 cols.
// Fragment layouts (ISA 7.12.2), per lane (half = lane>>4, m = lane&15):
//   A  : elems 0..7  -> K = k0 + 8*half + 0..7   (contiguous, one b128)
//        elems 8..15 -> K = k0 + 16 + 8*half + 0..7 (contiguous, one b128)
//   B  : same K order as A, column col0+m (two b128)
__global__ void gemm_wmma_kernel(const _Float16* __restrict__ Ah, const _Float16* __restrict__ Bt,
                                 const float* __restrict__ bias, const float* __restrict__ invd,
                                 float* __restrict__ Hout, float* __restrict__ Agg, int n) {
  const int row0 = blockIdx.x * 16;
  const int wave = threadIdx.x >> 5;
  const int lane = threadIdx.x & 31;
  const int col0 = wave * 16;
  const int half = lane >> 4;
  const int m    = lane & 15;

  const int r = min(row0 + m, n - 1);                 // clamped -> branchless loads
  const _Float16* arow = Ah + (size_t)r * F;
  const _Float16* brow = Bt + (size_t)(col0 + m) * F;

  v8f c = {};
  #pragma unroll
  for (int k0 = 0; k0 < F; k0 += 32) {
    v8h a0 = *(const v8h*)(arow + k0 + 8 * half);
    v8h a1 = *(const v8h*)(arow + k0 + 16 + 8 * half);
    v8h b0 = *(const v8h*)(brow + k0 + 8 * half);
    v8h b1 = *(const v8h*)(brow + k0 + 16 + 8 * half);
    v16h a = __builtin_shufflevector(a0, a1, 0,1,2,3,4,5,6,7,8,9,10,11,12,13,14,15);
    v16h b = __builtin_shufflevector(b0, b1, 0,1,2,3,4,5,6,7,8,9,10,11,12,13,14,15);
    c = __builtin_amdgcn_wmma_f32_16x16x32_f16(false, a, false, b, (short)0, c, false, false);
  }

  // C/D layout: VGPR rr -> M = rr + 8*half, N = m
  const float bv = bias[col0 + m];
  if (row0 + 16 <= n) {          // uniform fast path (always taken when n % 16 == 0)
    #pragma unroll
    for (int rr = 0; rr < 8; ++rr) {
      const int row = row0 + rr + 8 * half;
      const float hv = c[rr];
      Hout[(size_t)row * F + col0 + m] = hv;
      Agg [(size_t)row * F + col0 + m] = hv * invd[row] + bv;
    }
  } else {
    #pragma unroll
    for (int rr = 0; rr < 8; ++rr) {
      const int row = row0 + rr + 8 * half;
      if (row < n) {
        const float hv = c[rr];
        Hout[(size_t)row * F + col0 + m] = hv;
        Agg [(size_t)row * F + col0 + m] = hv * invd[row] + bv;
      }
    }
  }
}

// ---------------- edge scatter: agg[dst] += h[src] * isd[src]*isd[dst] ----------------
__global__ void edge_scatter_kernel(const float* __restrict__ h,
                                    const int* __restrict__ src, const int* __restrict__ dst,
                                    const float* __restrict__ isd,
                                    float* __restrict__ agg, int e) {
  int t = blockIdx.x * blockDim.x + threadIdx.x;
  int edge = t >> 5;
  if (edge >= e) return;
  int f = (t & 31) * 4;
  int s = src[edge], d = dst[edge];
  float norm = isd[s] * isd[d];
  float4 v = *(const float4*)(h + (size_t)s * F + f);
  float* p = agg + (size_t)d * F + f;
  atomicAdd(p + 0, v.x * norm);
  atomicAdd(p + 1, v.y * norm);
  atomicAdd(p + 2, v.z * norm);
  atomicAdd(p + 3, v.w * norm);
}

// ---------------- pooling (leaky fused on read) ----------------
__global__ void zero_pool_kernel(float* __restrict__ sums, float* __restrict__ cnt, int gF, int g) {
  int i = blockIdx.x * blockDim.x + threadIdx.x;
  if (i < gF) sums[i] = 0.0f;
  if (i < g)  cnt[i]  = 0.0f;
}

__global__ void pool_kernel(const float* __restrict__ agg, const int* __restrict__ bat,
                            float* __restrict__ sums, float* __restrict__ cnt, int n) {
  int t = blockIdx.x * blockDim.x + threadIdx.x;
  int node = t >> 5;
  if (node >= n) return;
  int f = (t & 31) * 4;
  int grp = bat[node];
  float4 v = *(const float4*)(agg + (size_t)node * F + f);
  v.x = v.x > 0.0f ? v.x : NEG_SLOPE * v.x;
  v.y = v.y > 0.0f ? v.y : NEG_SLOPE * v.y;
  v.z = v.z > 0.0f ? v.z : NEG_SLOPE * v.z;
  v.w = v.w > 0.0f ? v.w : NEG_SLOPE * v.w;
  float* p = sums + (size_t)grp * F + f;
  atomicAdd(p + 0, v.x);
  atomicAdd(p + 1, v.y);
  atomicAdd(p + 2, v.z);
  atomicAdd(p + 3, v.w);
  if ((t & 31) == 0) atomicAdd(&cnt[grp], 1.0f);
}

// ---------------- final FC: out[g] = mean(h_g) . fc_w + fc_b ----------------
__global__ void fc_kernel(const float* __restrict__ sums, const float* __restrict__ cnt,
                          const float* __restrict__ fcw, const float* __restrict__ fcb,
                          float* __restrict__ out) {
  __shared__ float red[F];
  int g = blockIdx.x, t = threadIdx.x;
  float c = fmaxf(cnt[g], 1.0f);
  red[t] = (sums[(size_t)g * F + t] / c) * fcw[t];
  __syncthreads();
  for (int s = F / 2; s > 0; s >>= 1) {
    if (t < s) red[t] += red[t + s];
    __syncthreads();
  }
  if (t == 0) out[g] = red[0] + fcb[0];
}

extern "C" void kernel_launch(void* const* d_in, const int* in_sizes, int n_in,
                              void* d_out, int out_size, void* d_ws, size_t ws_size,
                              hipStream_t stream) {
  const float* x   = (const float*)d_in[0];
  const int*   ei  = (const int*)  d_in[1];
  const int*   bat = (const int*)  d_in[2];
  const float* w1  = (const float*)d_in[3];
  const float* b1  = (const float*)d_in[4];
  const float* w2  = (const float*)d_in[5];
  const float* b2  = (const float*)d_in[6];
  const float* fcw = (const float*)d_in[7];
  const float* fcb = (const float*)d_in[8];
  float* out = (float*)d_out;

  const int n = in_sizes[0] / F;   // 100000 nodes
  const int e = in_sizes[1] / 2;   // 1600000 edges
  const int g = out_size;          // 64 graphs

  const int* src = ei;
  const int* dst = ei + e;

  // workspace carve-out (16B-aligned chunks)
  char* ws = (char*)d_ws;
  float*    deg  = (float*)ws;    ws += (size_t)n * 4;
  float*    isd  = (float*)ws;    ws += (size_t)n * 4;
  float*    invd = (float*)ws;    ws += (size_t)n * 4;
  float*    hbuf = (float*)ws;    ws += (size_t)n * F * 4;
  float*    agg  = (float*)ws;    ws += (size_t)n * F * 4;
  _Float16* xh   = (_Float16*)ws; ws += (size_t)n * F * 2;
  _Float16* w1t  = (_Float16*)ws; ws += (size_t)F * F * 2;
  _Float16* w2t  = (_Float16*)ws; ws += (size_t)F * F * 2;
  float*    sums = (float*)ws;    ws += (size_t)g * F * 4;
  float*    cnt  = (float*)ws;

  const int B  = 256;
  const int nF = n * F;

  // degrees + pool init
  init_deg_kernel<<<(n + B - 1) / B, B, 0, stream>>>(deg, n);
  count_deg_kernel<<<(e + B - 1) / B, B, 0, stream>>>(dst, deg, e);
  finalize_deg_kernel<<<(n + B - 1) / B, B, 0, stream>>>(deg, isd, invd, n);
  zero_pool_kernel<<<(g * F + B - 1) / B, B, 0, stream>>>(sums, cnt, g * F, g);

  // operand preparation for WMMA
  f32_to_f16_kernel<<<(nF + B - 1) / B, B, 0, stream>>>(x, xh, nF);
  convert_transpose_w_kernel<<<(F * F + B - 1) / B, B, 0, stream>>>(w1, w1t);
  convert_transpose_w_kernel<<<(F * F + B - 1) / B, B, 0, stream>>>(w2, w2t);

  const int gemm_blocks     = (n + 15) / 16;
  const int edge_thr_blocks = (int)(((long long)e * 32 + B - 1) / B);
  const int node_thr_blocks = (int)(((long long)n * 32 + B - 1) / B);

  // ---- layer 1 ----
  gemm_wmma_kernel<<<gemm_blocks, B, 0, stream>>>(xh, w1t, b1, invd, hbuf, agg, n);
  edge_scatter_kernel<<<edge_thr_blocks, B, 0, stream>>>(hbuf, src, dst, isd, agg, e);
  leaky_to_half_kernel<<<(nF + B - 1) / B, B, 0, stream>>>(agg, xh, nF);  // xh = act1 (f16)

  // ---- layer 2 ----
  gemm_wmma_kernel<<<gemm_blocks, B, 0, stream>>>(xh, w2t, b2, invd, hbuf, agg, n);
  edge_scatter_kernel<<<edge_thr_blocks, B, 0, stream>>>(hbuf, src, dst, isd, agg, e);

  // ---- pool (leaky fused) + FC ----
  pool_kernel<<<node_thr_blocks, B, 0, stream>>>(agg, bat, sums, cnt, n);
  fc_kernel<<<g, F, 0, stream>>>(sums, cnt, fcw, fcb, out);
}
